// QKVAttention_8942121910551
// MI455X (gfx1250) — matrix operand order repaired and checked by reading, not compiled, not run on hardware
//
#include <hip/hip_runtime.h>

typedef __attribute__((ext_vector_type(16))) __bf16 bf16x16;
typedef __attribute__((ext_vector_type(8)))  __bf16 bf16x8;
typedef __attribute__((ext_vector_type(4)))  __bf16 bf16x4;
typedef __attribute__((ext_vector_type(8)))  float  f32x8;
typedef __attribute__((ext_vector_type(4)))  float  f32x4;

constexpr int kB  = 16;
constexpr int kN  = 2048;
constexpr int kM  = 2048;
constexpr int kDK = 128;
constexpr int kDV = 128;
constexpr int QTILE  = 128;  // query rows per workgroup
constexpr int KTILE  = 32;   // keys per main-loop iteration
constexpr int NWAVES = 8;
constexpr int VSTR   = 40;   // padded Vt row stride (elements) -> 80B rows, 8B aligned

union BF16Frag {
    bf16x16 v;
    bf16x8  h[2];
    bf16x4  q[4];
};

__device__ __forceinline__ __bf16 f2bf(float f) {
    unsigned u = __float_as_uint(f);
    unsigned r = (u + 0x7FFFu + ((u >> 16) & 1u)) >> 16;   // round-to-nearest-even
    unsigned short s = (unsigned short)r;
    return __builtin_bit_cast(__bf16, s);
}

__global__ __launch_bounds__(256, 1)
void qkv_attn_fwd(const float* __restrict__ Q, const float* __restrict__ K,
                  const float* __restrict__ V, const float* __restrict__ pres,
                  float* __restrict__ Out)
{
    __shared__ __bf16 Qsh[QTILE][kDK];        // 32 KB
    __shared__ __bf16 Ksh[KTILE][kDK];        //  8 KB
    __shared__ __bf16 Vt [kDV][VSTR];         // 10 KB (transposed V tile)
    __shared__ __bf16 Psh[NWAVES][16][32];    //  8 KB (per-wave P scratch)

    const int tid  = threadIdx.x;
    const int lane = tid & 31;
    const int wave = tid >> 5;
    const int hi   = lane >> 4;   // 0/1: half-wave selector
    const int cl   = lane & 15;   // column within 16-group

    const int b  = blockIdx.y;
    const int q0 = blockIdx.x * QTILE;

    const float* Qb = Q    + (size_t)b * kN * kDK + (size_t)q0 * kDK;
    const float* Kb = K    + (size_t)b * kM * kDK;
    const float* Vb = V    + (size_t)b * kM * kDV;
    const float* Pb = pres + (size_t)b * kM;

    // per-thread staging coordinates (same for K and V tiles): 4 x f32x4 each
    const int sr = tid >> 5;          // key row handled this pass: f>>7 with f=(i*256+tid)*4
    const int sc = (tid & 31) * 4;    // dv/dk column base

    // ---- stage Q tile: 128x128 f32 -> bf16 LDS (coalesced f32x4 reads) ----
    #pragma unroll
    for (int i = 0; i < 16; ++i) {
        int f = (i * 256 + tid) * 4;
        int r = f >> 7, c = f & 127;
        f32x4 x = *(const f32x4*)(Qb + f);
        __bf16* dst = &Qsh[r][c];
        dst[0] = f2bf(x.x); dst[1] = f2bf(x.y); dst[2] = f2bf(x.z); dst[3] = f2bf(x.w);
    }

    // ---- prefetch K/V tile 0 into registers (fp32), presence too ----
    f32x4 kpre[4], vpre[4];
    #pragma unroll
    for (int i = 0; i < 4; ++i) {
        kpre[i] = *(const f32x4*)(Kb + (size_t)(i * 8 + sr) * kDK + sc);
        vpre[i] = *(const f32x4*)(Vb + (size_t)(i * 8 + sr) * kDV + sc);
    }
    float presA = Pb[cl];
    float presB = Pb[16 + cl];

    __syncthreads();

    // ---- per-wave Q A-fragments, kept in registers for whole kernel ----
    BF16Frag qf[4];
    {
        const __bf16* qrow = &Qsh[wave * 16 + cl][0];
        #pragma unroll
        for (int dc = 0; dc < 4; ++dc) {
            int kb = dc * 32 + hi * 16;
            qf[dc].h[0] = *(const bf16x8*)(qrow + kb);       // K = kb..kb+7
            qf[dc].h[1] = *(const bf16x8*)(qrow + kb + 8);   // K = kb+8..kb+15
        }
    }

    // ---- commit tile 0 to LDS ----
    #pragma unroll
    for (int i = 0; i < 4; ++i) {
        int r = i * 8 + sr;
        __bf16* kd = &Ksh[r][sc];
        kd[0] = f2bf(kpre[i].x); kd[1] = f2bf(kpre[i].y);
        kd[2] = f2bf(kpre[i].z); kd[3] = f2bf(kpre[i].w);
        Vt[sc + 0][r] = f2bf(vpre[i].x);
        Vt[sc + 1][r] = f2bf(vpre[i].y);
        Vt[sc + 2][r] = f2bf(vpre[i].z);
        Vt[sc + 3][r] = f2bf(vpre[i].w);
    }
    __syncthreads();

    f32x8 acc[8] = {};          // 16x128 output accumulator (8 chunks of 16 cols)
    float mrow[8], lrow[8];     // per-row running max / sum (rows j + 8*hi)
    #pragma unroll
    for (int j = 0; j < 8; ++j) { mrow[j] = -1e30f; lrow[j] = 0.0f; }

    const float SC = 0.1275174245f;   // (1/sqrt(128)) * log2(e)

    for (int kt = 0; kt < kM; kt += KTILE) {
        const bool more = (kt + KTILE) < kM;

        // ---- issue global prefetch for tile kt+32 early (hidden behind WMMAs) ----
        float npresA = 0.0f, npresB = 0.0f;
        if (more) {
            const float* Kn = Kb + (size_t)(kt + KTILE) * kDK;
            const float* Vn = Vb + (size_t)(kt + KTILE) * kDV;
            #pragma unroll
            for (int i = 0; i < 4; ++i) {
                kpre[i] = *(const f32x4*)(Kn + (size_t)(i * 8 + sr) * kDK + sc);
                vpre[i] = *(const f32x4*)(Vn + (size_t)(i * 8 + sr) * kDV + sc);
            }
            npresA = Pb[kt + KTILE + cl];
            npresB = Pb[kt + KTILE + 16 + cl];
        }

        // ---- S = Q K^T : two 16x16 tiles (keys 0-15, 16-31), K-dim = 128 ----
        f32x8 Sa = {}, Sb = {};
        #pragma unroll
        for (int dc = 0; dc < 4; ++dc) {
            BF16Frag ka, kb2;
            int kbase = dc * 32 + hi * 16;   // B frag: contiguous K run per lane
            ka.h[0]  = *(const bf16x8*)(&Ksh[cl][kbase]);
            ka.h[1]  = *(const bf16x8*)(&Ksh[cl][kbase + 8]);
            kb2.h[0] = *(const bf16x8*)(&Ksh[16 + cl][kbase]);
            kb2.h[1] = *(const bf16x8*)(&Ksh[16 + cl][kbase + 8]);
            Sa = __builtin_amdgcn_wmma_f32_16x16x32_bf16(false, qf[dc].v, false, ka.v,
                                                         (short)0, Sa, false, false);
            Sb = __builtin_amdgcn_wmma_f32_16x16x32_bf16(false, qf[dc].v, false, kb2.v,
                                                         (short)0, Sb, false, false);
        }

        // ---- mask + online softmax (row-wise over keys) ----
        #pragma unroll
        for (int j = 0; j < 8; ++j) {
            float sa = (presA != 0.0f) ? Sa[j] * SC : -1e30f;
            float sb = (presB != 0.0f) ? Sb[j] * SC : -1e30f;
            float t = fmaxf(sa, sb);
            #pragma unroll
            for (int o = 1; o < 16; o <<= 1) t = fmaxf(t, __shfl_xor(t, o, 16));
            float mnew = fmaxf(mrow[j], t);
            float corr = exp2f(mrow[j] - mnew);
            mrow[j] = mnew;
            float pa = exp2f(sa - mnew);
            float pb = exp2f(sb - mnew);
            float rs = pa + pb;
            #pragma unroll
            for (int o = 1; o < 16; o <<= 1) rs += __shfl_xor(rs, o, 16);
            lrow[j] = lrow[j] * corr + rs;
            #pragma unroll
            for (int c = 0; c < 8; ++c) acc[c][j] *= corr;
            // C-layout -> LDS so we can re-read P in A-layout (same-wave DS is in-order)
            Psh[wave][j + 8 * hi][cl]      = f2bf(pa);
            Psh[wave][j + 8 * hi][16 + cl] = f2bf(pb);
        }

        // ---- O += P (16x32) x V (32x128) ----
        BF16Frag pf;
        const __bf16* prow = &Psh[wave][cl][0];
        pf.h[0] = *(const bf16x8*)(prow + hi * 16);
        pf.h[1] = *(const bf16x8*)(prow + hi * 16 + 8);
        #pragma unroll
        for (int c = 0; c < 8; ++c) {
            BF16Frag vf;
            const __bf16* vrow = &Vt[c * 16 + cl][hi * 16];
            vf.q[0] = *(const bf16x4*)(vrow + 0);
            vf.q[1] = *(const bf16x4*)(vrow + 4);
            vf.q[2] = *(const bf16x4*)(vrow + 8);
            vf.q[3] = *(const bf16x4*)(vrow + 12);
            acc[c] = __builtin_amdgcn_wmma_f32_16x16x32_bf16(false, pf.v, false, vf.v,
                                                             (short)0, acc[c], false, false);
        }

        // ---- commit prefetched tile to LDS for next iteration ----
        if (more) {
            __syncthreads();   // all waves done reading current Ksh/Vt
            #pragma unroll
            for (int i = 0; i < 4; ++i) {
                int r = i * 8 + sr;
                __bf16* kd = &Ksh[r][sc];
                kd[0] = f2bf(kpre[i].x); kd[1] = f2bf(kpre[i].y);
                kd[2] = f2bf(kpre[i].z); kd[3] = f2bf(kpre[i].w);
                Vt[sc + 0][r] = f2bf(vpre[i].x);
                Vt[sc + 1][r] = f2bf(vpre[i].y);
                Vt[sc + 2][r] = f2bf(vpre[i].z);
                Vt[sc + 3][r] = f2bf(vpre[i].w);
            }
            __syncthreads();   // tile visible to all waves
            presA = npresA;
            presB = npresB;
        }
    }

    // ---- epilogue: normalize by running sum, write fp32 output ----
    float* Ob = Out + (size_t)b * kN * kDV + (size_t)(q0 + wave * 16) * kDV;
    #pragma unroll
    for (int j = 0; j < 8; ++j) {
        float inv = 1.0f / lrow[j];
        int r = j + 8 * hi;
        #pragma unroll
        for (int c = 0; c < 8; ++c) {
            Ob[(size_t)r * kDV + c * 16 + cl] = acc[c][j] * inv;
        }
    }
}

extern "C" void kernel_launch(void* const* d_in, const int* in_sizes, int n_in,
                              void* d_out, int out_size, void* d_ws, size_t ws_size,
                              hipStream_t stream) {
    (void)in_sizes; (void)n_in; (void)out_size; (void)d_ws; (void)ws_size;
    const float* Q = (const float*)d_in[0];
    const float* K = (const float*)d_in[1];
    const float* V = (const float*)d_in[2];
    const float* P = (const float*)d_in[3];
    float* O = (float*)d_out;
    dim3 grid(kN / QTILE, kB, 1);
    qkv_attn_fwd<<<grid, dim3(256, 1, 1), 0, stream>>>(Q, K, V, P, O);
}
